// SelfAttention_82471962017999
// MI455X (gfx1250) — matrix operand order repaired and checked by reading, not compiled, not run on hardware
//
#include <hip/hip_runtime.h>
#include <hip/hip_bf16.h>
#include <cstdint>

// ---------------------------------------------------------------------------
// MI455X (gfx1250) self-attention, flash-attention style, all matmuls via
// v_wmma_f32_16x16x32_f16 (f16 inputs, f32 accumulate). Attention K/V feed
// uses the Tensor Data Mover (tensor_load_to_lds) with double buffering,
// gated by s_wait_tensorcnt.
// ---------------------------------------------------------------------------

typedef __attribute__((ext_vector_type(16))) _Float16     v16h;
typedef __attribute__((ext_vector_type(8)))  float        v8f;
typedef __attribute__((ext_vector_type(4)))  unsigned int v4u;
typedef __attribute__((ext_vector_type(4)))  unsigned int u32x4;
typedef __attribute__((ext_vector_type(8)))  int          i32x8;
typedef __attribute__((ext_vector_type(4)))  int          i32x4;

union Frag16 { v16h h; v4u u[2]; };

constexpr int SEQ   = 4096;
constexpr int EMB   = 768;
constexpr int NHEAD = 12;
constexpr int KT32  = EMB / 32;   // 24 K-chunks of 32
constexpr int NT16  = EMB / 16;   // 48 N-tiles of 16
constexpr int MT16  = SEQ / 16;   // 256 M-tiles of 16

static __device__ __forceinline__ v8f wmma_f16(v16h a, v16h b, v8f c) {
  // D = A(16x32 f16) * B(32x16 f16) + C(16x16 f32)
  return __builtin_amdgcn_wmma_f32_16x16x32_f16(
      /*neg_a=*/false, a, /*neg_b=*/false, b,
      /*c_mod=*/(short)0, c, /*reuse_a=*/false, /*reuse_b=*/false);
}

// ---------------------------------------------------------------------------
// TDM: DMA one contiguous row of `elems` 2-byte elements from global memory
// into LDS at byte offset lds_off. Descriptor per CDNA5 ISA ch.8 (D# groups):
//   group0: count=1 | lds_addr | global_addr[56:0] | type=2
//   group1: data_size=1 (2B), tensor_dim0 = tile_dim0 = elems, tile_dim1 = 1,
//           tensor_dim0_stride = elems, no padding / iterate / multicast.
// ---------------------------------------------------------------------------
static __device__ __forceinline__ void tdm_load_row_f16(unsigned lds_off,
                                                        const void* gptr,
                                                        unsigned elems) {
  const unsigned long long ga = (unsigned long long)(uintptr_t)gptr;
  u32x4 g0;
  g0[0] = 1u;                                            // count=1 (valid)
  g0[1] = lds_off;                                       // lds_addr (bytes)
  g0[2] = (unsigned)(ga & 0xffffffffu);                  // global_addr[31:0]
  g0[3] = (unsigned)((ga >> 32) & 0x01ffffffu) | 0x80000000u; // [56:32]|type=2
  i32x8 g1;
  g1[0] = (int)(1u << 16);                               // data_size=1 -> 2B
  g1[1] = (int)((elems & 0xffffu) << 16);                // tensor_dim0[15:0]
  g1[2] = (int)(((elems >> 16) & 0xffffu) | (1u << 16)); // dim0[31:16],dim1=1
  g1[3] = (int)((elems & 0xffffu) << 16);                // tile_dim0
  g1[4] = 1;                                             // tile_dim1=1
  g1[5] = (int)elems;                                    // dim0_stride[31:0]
  g1[6] = 0;
  g1[7] = 0;
  const i32x4 z4 = (i32x4)0;
#if defined(__clang_major__) && (__clang_major__ >= 23)
  const i32x8 z8 = (i32x8)0;
  __builtin_amdgcn_tensor_load_to_lds(g0, g1, z4, z4, z8, 0);
#else
  __builtin_amdgcn_tensor_load_to_lds(g0, g1, z4, z4, 0);
#endif
}

// ---------------------------------------------------------------------------
// Pack fp32 [M x K] row-major into A-fragment layout:
//   tile (mt, kt) of 16x32 -> 512 halves; lane l owns row M = l%16 and
//   halves e=0..7 -> K = (l>=16?8:0)+e ; e=8..15 -> K = 16+(l>=16?8:0)+(e-8)
// stored contiguously at lane*16 so the GEMM loads 32B per lane.
// ---------------------------------------------------------------------------
__global__ __launch_bounds__(32)
void pack_a_kernel(const float* __restrict__ src, _Float16* __restrict__ dst,
                   int ldk, int ktiles) {
  const int mt = blockIdx.x, kt = blockIdx.y, l = threadIdx.x;
  const int m  = mt * 16 + (l & 15);
  const int kb = kt * 32 + ((l >> 4) * 8);
  const float* p = src + (size_t)m * ldk + kb;
  Frag16 f;
#pragma unroll
  for (int e = 0; e < 8; ++e) f.h[e]     = (_Float16)p[e];
#pragma unroll
  for (int e = 0; e < 8; ++e) f.h[8 + e] = (_Float16)p[16 + e];
  *(v16h*)(dst + ((size_t)mt * ktiles + kt) * 512 + l * 16) = f.h;
}

// ---------------------------------------------------------------------------
// Pack fp32 W [K x N] (row-major, K=in, N=out) into B-fragment layout:
//   tile (kt, nt) of 32x16 stored at n*32+16h+e (h=l>>4), K=8h+e (e<8), K=8+8h+e (e>=8); the GEMM's lane l
//   reads 16 contiguous halves at (l%16)*32 + (l>>4)*16.
// ---------------------------------------------------------------------------
__global__ __launch_bounds__(32)
void pack_b_kernel(const float* __restrict__ W, _Float16* __restrict__ dst) {
  const int kt = blockIdx.x, nt = blockIdx.y, l = threadIdx.x;
  const int n  = l & 15;
  const int k0 = (l >> 4) * 16;
  Frag16 f;
#pragma unroll
  for (int i = 0; i < 16; ++i)
    f.h[i] = (_Float16)W[(size_t)(kt * 32 + (k0 >> 1) + (i & 7) + (i >> 3) * 16) * EMB + nt * 16 + n];
  *(v16h*)(dst + ((size_t)kt * NT16 + nt) * 512 + n * 32 + k0) = f.h;
}

// ---------------------------------------------------------------------------
// WMMA GEMM: one wave computes a 32x32 output block of  A(4096x768)@B(768xN).
// Epilogue modes:
//   0: fp32 row-major out + bias (final projection)
//   1: Q   -> A-fragment pack  [head][mtile][kc(2)][512]
//   2: K^T -> B-fragment pack  [head][keytile(256)][kc(2)][512]
//   3: V   -> B-fragment pack  [head][keychunk(128)][ntile(4)][512]
// ---------------------------------------------------------------------------
__global__ __launch_bounds__(32)
void gemm_wmma_kernel(const _Float16* __restrict__ Apk,
                      const _Float16* __restrict__ Bpk,
                      const float* __restrict__ bias,
                      float* __restrict__ outF,
                      _Float16* __restrict__ outP,
                      int ktiles, int mode) {
  __shared__ __align__(16) _Float16 sm[32 * 32];
  const int bx = blockIdx.x, by = blockIdx.y, l = threadIdx.x;
  const int lh = l >> 4, ln = l & 15;
  const int boff = ln * 32 + lh * 16;
  v8f a00 = {}, a01 = {}, a10 = {}, a11 = {};
  const size_t arow0 = ((size_t)(2 * bx)     * ktiles) * 512;
  const size_t arow1 = ((size_t)(2 * bx + 1) * ktiles) * 512;
  const size_t bcol0 = (size_t)(2 * by)     * 512;
  const size_t bcol1 = (size_t)(2 * by + 1) * 512;

  for (int kt = 0; kt < ktiles; ++kt) {
    Frag16 fa0, fa1, fb0, fb1;
    fa0.h = *(const v16h*)(Apk + arow0 + (size_t)kt * 512 + l * 16);
    fa1.h = *(const v16h*)(Apk + arow1 + (size_t)kt * 512 + l * 16);
    fb0.h = *(const v16h*)(Bpk + (size_t)kt * NT16 * 512 + bcol0 + boff);
    fb1.h = *(const v16h*)(Bpk + (size_t)kt * NT16 * 512 + bcol1 + boff);
    if (kt + 1 < ktiles) {  // global_prefetch_b8 of next K-step
      __builtin_prefetch(Apk + arow0 + (size_t)(kt + 1) * 512 + l * 16, 0, 1);
      __builtin_prefetch(Bpk + (size_t)(kt + 1) * NT16 * 512 + bcol0 + boff, 0, 1);
    }
    a00 = wmma_f16(fa0.h, fb0.h, a00);
    a01 = wmma_f16(fa0.h, fb1.h, a01);
    a10 = wmma_f16(fa1.h, fb0.h, a10);
    a11 = wmma_f16(fa1.h, fb1.h, a11);
  }

  const int n0 = by * 32, m0 = bx * 32;
  const float b0v = bias[n0 + ln];
  const float b1v = bias[n0 + 16 + ln];

  if (mode == 0) {  // fp32 row-major + bias
#pragma unroll
    for (int r = 0; r < 8; ++r) {
      const int row0 = m0 + r + 8 * lh;
      const int row1 = row0 + 16;
      outF[(size_t)row0 * EMB + n0 + ln]      = a00[r] + b0v;
      outF[(size_t)row0 * EMB + n0 + 16 + ln] = a01[r] + b1v;
      outF[(size_t)row1 * EMB + n0 + ln]      = a10[r] + b0v;
      outF[(size_t)row1 * EMB + n0 + 16 + ln] = a11[r] + b1v;
    }
    return;
  }

  // Stage 32x32 f16 block to LDS. mode 3 (V) stages transposed [N][M] so the
  // B-fragment read (16 contiguous K=key halves per lane) is contiguous.
  if (mode == 3) {
#pragma unroll
    for (int r = 0; r < 8; ++r) {
      const int mr0 = r + 8 * lh, mr1 = mr0 + 16;
      sm[ln * 32 + mr0]        = (_Float16)(a00[r] + b0v);
      sm[(16 + ln) * 32 + mr0] = (_Float16)(a01[r] + b1v);
      sm[ln * 32 + mr1]        = (_Float16)(a10[r] + b0v);
      sm[(16 + ln) * 32 + mr1] = (_Float16)(a11[r] + b1v);
    }
  } else {
#pragma unroll
    for (int r = 0; r < 8; ++r) {
      const int mr0 = r + 8 * lh, mr1 = mr0 + 16;
      sm[mr0 * 32 + ln]      = (_Float16)(a00[r] + b0v);
      sm[mr0 * 32 + 16 + ln] = (_Float16)(a01[r] + b1v);
      sm[mr1 * 32 + ln]      = (_Float16)(a10[r] + b0v);
      sm[mr1 * 32 + 16 + ln] = (_Float16)(a11[r] + b1v);
    }
  }
  __syncthreads();

  const int head = n0 >> 6;
  if (mode == 1) {            // Q: A-fragment pack per head
    const int kc = (n0 >> 5) & 1;
#pragma unroll
    for (int i = 0; i < 2; ++i) {
      const int a = (i * 16 + ln) * 32 + lh * 8;
      v4u u0 = *(const v4u*)(sm + a);
      v4u u1 = *(const v4u*)(sm + a + 16);
      _Float16* d = outP + (((size_t)head * MT16 + (2 * bx + i)) * 2 + kc) * 512 + l * 16;
      *(v4u*)d = u0; *(v4u*)(d + 8) = u1;
    }
  } else if (mode == 2) {     // K^T: B-fragment pack (K of WMMA = head dims)
    const int kc = (n0 >> 5) & 1;
#pragma unroll
    for (int i = 0; i < 2; ++i) {
      const int a = (i * 16 + ln) * 32 + lh * 8;
      v4u u0 = *(const v4u*)(sm + a);
      v4u u1 = *(const v4u*)(sm + a + 16);
      _Float16* d = outP + (((size_t)head * MT16 + (2 * bx + i)) * 2 + kc) * 512
                    + ln * 32 + lh * 16;
      *(v4u*)d = u0; *(v4u*)(d + 8) = u1;
    }
  } else {                    // V: B-fragment pack (K of WMMA = key positions)
    const int ntih = (n0 & 63) >> 4;
#pragma unroll
    for (int j = 0; j < 2; ++j) {
      const int a = (j * 16 + ln) * 32 + lh * 8;
      v4u u0 = *(const v4u*)(sm + a);
      v4u u1 = *(const v4u*)(sm + a + 16);
      _Float16* d = outP + (((size_t)head * (MT16 / 2) + bx) * 4 + ntih + j) * 512
                    + ln * 32 + lh * 16;
      *(v4u*)d = u0; *(v4u*)(d + 8) = u1;
    }
  }
}

// ---------------------------------------------------------------------------
// Flash attention: one wave per (head, 16-query tile). 32 keys / iteration.
// K^T and V B-fragments for each iteration are two contiguous 4KB blocks;
// they are DMA'd into LDS by the Tensor Data Mover, double buffered, gated
// with s_wait_tensorcnt. Per iteration: 4 WMMAs for S=Q*K^T, fp32 online
// softmax (shfl_xor row reductions), P transposed C->A layout via LDS,
// 4 WMMAs for O += P*V. Output written in A-fragment layout.
// ---------------------------------------------------------------------------
__global__ __launch_bounds__(32)
void attn_kernel(const _Float16* __restrict__ Qp, const _Float16* __restrict__ KTp,
                 const _Float16* __restrict__ Vp, _Float16* __restrict__ Op) {
  __shared__ __align__(16) _Float16 kv[2][4096];  // [buf][K(2048) | V(2048)]
  __shared__ __align__(16) _Float16 sm[16 * 64];
  const int mt = blockIdx.x, h = blockIdx.y, l = threadIdx.x;
  const int lh = l >> 4, ln = l & 15;
  const int boff = ln * 32 + lh * 16;
  const float scale = 0.03608439182435161f;  // 1/sqrt(768), per reference

  const unsigned kvoff[2] = { (unsigned)(uintptr_t)&kv[0][0],
                              (unsigned)(uintptr_t)&kv[1][0] };

  Frag16 qa0, qa1;
  qa0.h = *(const v16h*)(Qp + (((size_t)h * MT16 + mt) * 2 + 0) * 512 + l * 16);
  qa1.h = *(const v16h*)(Qp + (((size_t)h * MT16 + mt) * 2 + 1) * 512 + l * 16);

  v8f o0 = {}, o1 = {}, o2 = {}, o3 = {};
  float mrun[8], lrun[8];
#pragma unroll
  for (int r = 0; r < 8; ++r) { mrun[r] = -1e30f; lrun[r] = 0.f; }

  auto issue_kv = [&](int kt) {
    const unsigned dst = kvoff[kt & 1];
    tdm_load_row_f16(dst,
                     KTp + (((size_t)h * MT16 + 2 * kt) * 2) * 512, 2048);
    tdm_load_row_f16(dst + 4096,
                     Vp + (((size_t)h * (SEQ / 32) + kt) * 4) * 512, 2048);
  };
  issue_kv(0);

  for (int kt = 0; kt < SEQ / 32; ++kt) {
    if (kt + 1 < SEQ / 32) {
      issue_kv(kt + 1);                       // prefetch next K/V pair via TDM
      __builtin_amdgcn_s_wait_tensorcnt(2);   // current buffer's 2 ops done
    } else {
      __builtin_amdgcn_s_wait_tensorcnt(0);
    }
    asm volatile("" ::: "memory");
    const _Float16* kvb = kv[kt & 1];

    Frag16 kb00, kb01, kb10, kb11;
    kb00.h = *(const v16h*)(kvb + 0 * 512 + boff);
    kb01.h = *(const v16h*)(kvb + 1 * 512 + boff);
    kb10.h = *(const v16h*)(kvb + 2 * 512 + boff);
    kb11.h = *(const v16h*)(kvb + 3 * 512 + boff);
    v8f s0 = {}, s1 = {};
    s0 = wmma_f16(qa0.h, kb00.h, s0);
    s0 = wmma_f16(qa1.h, kb01.h, s0);
    s1 = wmma_f16(qa0.h, kb10.h, s1);
    s1 = wmma_f16(qa1.h, kb11.h, s1);

#pragma unroll
    for (int r = 0; r < 8; ++r) {
      const float x0 = s0[r] * scale, x1 = s1[r] * scale;
      float mx = fmaxf(x0, x1);
      mx = fmaxf(mx, __shfl_xor(mx, 1));
      mx = fmaxf(mx, __shfl_xor(mx, 2));
      mx = fmaxf(mx, __shfl_xor(mx, 4));
      mx = fmaxf(mx, __shfl_xor(mx, 8));
      const float mnew  = fmaxf(mrun[r], mx);
      const float alpha = __expf(mrun[r] - mnew);
      const float p0 = __expf(x0 - mnew);
      const float p1 = __expf(x1 - mnew);
      float rs = p0 + p1;
      rs += __shfl_xor(rs, 1);
      rs += __shfl_xor(rs, 2);
      rs += __shfl_xor(rs, 4);
      rs += __shfl_xor(rs, 8);
      lrun[r] = lrun[r] * alpha + rs;
      mrun[r] = mnew;
      o0[r] = o0[r] * alpha; o1[r] = o1[r] * alpha;
      o2[r] = o2[r] * alpha; o3[r] = o3[r] * alpha;
      const int mrow = r + 8 * lh;
      sm[mrow * 32 + ln]      = (_Float16)p0;  // [M][K] f16 P tile
      sm[mrow * 32 + 16 + ln] = (_Float16)p1;
    }
    __syncthreads();

    Frag16 pa;  // reload P in A-fragment layout (two contiguous b128 chunks)
    const int a = ln * 32 + lh * 8;
    pa.u[0] = *(const v4u*)(sm + a);
    pa.u[1] = *(const v4u*)(sm + a + 16);

    Frag16 vb0, vb1, vb2, vb3;
    vb0.h = *(const v16h*)(kvb + 2048 + 0 * 512 + boff);
    vb1.h = *(const v16h*)(kvb + 2048 + 1 * 512 + boff);
    vb2.h = *(const v16h*)(kvb + 2048 + 2 * 512 + boff);
    vb3.h = *(const v16h*)(kvb + 2048 + 3 * 512 + boff);
    o0 = wmma_f16(pa.h, vb0.h, o0);
    o1 = wmma_f16(pa.h, vb1.h, o1);
    o2 = wmma_f16(pa.h, vb2.h, o2);
    o3 = wmma_f16(pa.h, vb3.h, o3);
    __syncthreads();
  }

  // Normalize and stage 16x64 O tile, then emit A-fragment layout.
#pragma unroll
  for (int r = 0; r < 8; ++r) {
    const float inv = 1.f / lrun[r];
    const int mrow = r + 8 * lh;
    sm[mrow * 64 + ln]      = (_Float16)(o0[r] * inv);
    sm[mrow * 64 + 16 + ln] = (_Float16)(o1[r] * inv);
    sm[mrow * 64 + 32 + ln] = (_Float16)(o2[r] * inv);
    sm[mrow * 64 + 48 + ln] = (_Float16)(o3[r] * inv);
  }
  __syncthreads();
#pragma unroll
  for (int kc = 0; kc < 2; ++kc) {
    const int a = ln * 64 + kc * 32 + lh * 8;
    v4u u0 = *(const v4u*)(sm + a);
    v4u u1 = *(const v4u*)(sm + a + 16);
    _Float16* d = Op + ((size_t)mt * KT32 + (h * 2 + kc)) * 512 + l * 16;
    *(v4u*)d = u0; *(v4u*)(d + 8) = u1;
  }
}

// ---------------------------------------------------------------------------
extern "C" void kernel_launch(void* const* d_in, const int* in_sizes, int n_in,
                              void* d_out, int out_size, void* d_ws, size_t ws_size,
                              hipStream_t stream) {
  const float* x  = (const float*)d_in[0];
  const float* Wq = (const float*)d_in[1];
  const float* bq = (const float*)d_in[2];
  const float* Wk = (const float*)d_in[3];
  const float* bk = (const float*)d_in[4];
  const float* Wv = (const float*)d_in[5];
  const float* bv = (const float*)d_in[6];
  const float* Wo = (const float*)d_in[7];
  const float* bo = (const float*)d_in[8];
  float* out = (float*)d_out;

  char* ws = (char*)d_ws;
  const size_t SZ_X = (size_t)SEQ * EMB * sizeof(_Float16);   // 6 MB
  const size_t SZ_W = (size_t)EMB * EMB * sizeof(_Float16);   // 1.125 MB
  _Float16* XH  = (_Float16*)(ws);
  _Float16* WQp = (_Float16*)(ws + SZ_X);
  _Float16* WKp = (_Float16*)(ws + SZ_X + 1 * SZ_W);
  _Float16* WVp = (_Float16*)(ws + SZ_X + 2 * SZ_W);
  _Float16* WOp = (_Float16*)(ws + SZ_X + 3 * SZ_W);
  _Float16* QP  = (_Float16*)(ws + 1 * SZ_X + 4 * SZ_W);
  _Float16* KTP = (_Float16*)(ws + 2 * SZ_X + 4 * SZ_W);
  _Float16* VP  = (_Float16*)(ws + 3 * SZ_X + 4 * SZ_W);
  _Float16* ATP = (_Float16*)(ws + 4 * SZ_X + 4 * SZ_W);   // ~34.5 MB total

  const dim3 blk(32);
  pack_a_kernel<<<dim3(MT16, KT32), blk, 0, stream>>>(x, XH, EMB, KT32);
  pack_b_kernel<<<dim3(KT32, NT16), blk, 0, stream>>>(Wq, WQp);
  pack_b_kernel<<<dim3(KT32, NT16), blk, 0, stream>>>(Wk, WKp);
  pack_b_kernel<<<dim3(KT32, NT16), blk, 0, stream>>>(Wv, WVp);
  pack_b_kernel<<<dim3(KT32, NT16), blk, 0, stream>>>(Wo, WOp);

  const dim3 gg(SEQ / 32, EMB / 32);
  gemm_wmma_kernel<<<gg, blk, 0, stream>>>(XH, WQp, bq, nullptr, QP,  KT32, 1);
  gemm_wmma_kernel<<<gg, blk, 0, stream>>>(XH, WKp, bk, nullptr, KTP, KT32, 2);
  gemm_wmma_kernel<<<gg, blk, 0, stream>>>(XH, WVp, bv, nullptr, VP,  KT32, 3);

  attn_kernel<<<dim3(MT16, NHEAD), blk, 0, stream>>>(QP, KTP, VP, ATP);

  gemm_wmma_kernel<<<gg, blk, 0, stream>>>(ATP, WOp, bo, out, nullptr, KT32, 0);
}
